// DiscreteDiT_31653908971776
// MI455X (gfx1250) — matrix operand order repaired and checked by reading, not compiled, not run on hardware
//
#include <hip/hip_runtime.h>
#include <hip/hip_bf16.h>
#include <math.h>

// ---------------------------------------------------------------------------
// DiT forward (B=4, N=192, D=256, H=4, HD=64, L=4) + pairwise MLP + Sinkhorn.
// - GEMMs: v_wmma_f32_16x16x32_bf16, fp32 accumulate; weights repacked once
//   per launch into the native per-lane B-fragment layout (2x b128 loads per
//   WMMA instead of 16x strided b32).
// - Pairwise stage fused in LDS; h tile staged via the Tensor Data Mover
//   (tensor_load_to_lds + s_wait_tensorcnt, 6-arg toolchain form).
// - Activations via fast v_exp/v_log/v_rcp paths.
// ---------------------------------------------------------------------------

#define NTOK 192
#define DMODEL 256
#define NHEAD 4
#define HDIM 64
#define NLAYER 4
#define BATCH 4
#define BN (BATCH * NTOK)        // 768 rows
#define INPUT_DIM 1728
#define INV_TEMP 20.0f           // 1 / 0.05
#define SINK_ITERS 20

typedef __attribute__((ext_vector_type(16))) __bf16    v16bf;
typedef __attribute__((ext_vector_type(8)))  float     v8f;
typedef __attribute__((ext_vector_type(8)))  unsigned  v8u;
typedef __attribute__((ext_vector_type(4)))  unsigned  v4u;
typedef __attribute__((ext_vector_type(8)))  int       v8i;
typedef __attribute__((ext_vector_type(4)))  int       v4i;

// pack two f32 into two bf16 (truncate) with one v_perm_b32
static __device__ __forceinline__ unsigned pack_bf16(float lo, float hi) {
  return __builtin_amdgcn_perm(__builtin_bit_cast(unsigned, hi),
                               __builtin_bit_cast(unsigned, lo), 0x07060302u);
}
static __device__ __forceinline__ float fast_rcp(float x) {
  return __builtin_amdgcn_rcpf(x);
}
// gelu(x) = x * sigmoid(2*0.79788456*(x + 0.044715 x^3))
static __device__ __forceinline__ float gelu_f(float x) {
  float t = 1.5957691216f * (x + 0.044715f * x * x * x);
  float e = __expf(t);
  return x * e * fast_rcp(e + 1.0f);
}
static __device__ __forceinline__ float silu_f(float x) {
  return x * fast_rcp(1.0f + __expf(-x));
}

// A fragment (ISA 16-bit A 16x32): p = &A[row*lda + kk + 8*half]
static __device__ __forceinline__ v16bf load_afrag(const float* __restrict__ p) {
  v8u a;
#pragma unroll
  for (int v = 0; v < 4; ++v) {
    float2 x = *(const float2*)(p + 2 * v);
    float2 y = *(const float2*)(p + 16 + 2 * v);
    a[v]     = pack_bf16(x.x, x.y);
    a[4 + v] = pack_bf16(y.x, y.y);
  }
  return __builtin_bit_cast(v16bf, a);
}

// Pre-packed B fragment: PB[((nt*ktiles + kk32)*32 + lane)*8 .. +7]
static __device__ __forceinline__ v16bf load_bpacked(const unsigned* __restrict__ PB,
                                                     int ktiles, int nt, int kk32,
                                                     int lane) {
  const uint4* p = (const uint4*)(PB + (((size_t)nt * ktiles + kk32) * 32 + lane) * 8);
  uint4 lo = p[0], hi = p[1];
  v8u r = {lo.x, lo.y, lo.z, lo.w, hi.x, hi.y, hi.z, hi.w};
  return __builtin_bit_cast(v16bf, r);
}

// ---------------------------------------------------------------------------
// Repack a K x N fp32 weight into bf16 WMMA-native B-fragment layout.
// One thread produces the 8 dwords for one (ntile, ktile, lane).
// ---------------------------------------------------------------------------
__global__ __launch_bounds__(256) void repack_kernel(
    const float* __restrict__ Wm, unsigned* __restrict__ PB, int K, int N) {
  int idx = blockIdx.x * 256 + threadIdx.x;
  int ktiles = K >> 5;
  int total = (N >> 4) * ktiles * 32;
  if (idx >= total) return;
  int lane = idx & 31;
  int kk32 = (idx >> 5) % ktiles;
  int nt   = (idx >> 5) / ktiles;
  int half = lane >> 4, ml = lane & 15;
  const float* p = Wm + (size_t)(kk32 * 32 + 8 * half) * N + nt * 16 + ml;
  unsigned o[8];
#pragma unroll
  for (int v = 0; v < 8; ++v)
    o[v] = pack_bf16(p[(size_t)(2 * v + (v < 4 ? 0 : 8)) * N], p[(size_t)(2 * v + (v < 4 ? 0 : 8) + 1) * N]);
  uint4* dst = (uint4*)(PB + (size_t)idx * 8);
  dst[0] = make_uint4(o[0], o[1], o[2], o[3]);
  dst[1] = make_uint4(o[4], o[5], o[6], o[7]);
}

// ---------------------------------------------------------------------------
// Single-wave WMMA GEMM: 16x32 output strip per wave (2 N-tiles share each A
// fragment), K unrolled x2 -> 4 independent accumulators.
// EPI: 0 = bias, 1 = bias+gelu, 2 = gated residual C += gate[b,col]*(acc+bias)
// ---------------------------------------------------------------------------
template <int EPI>
__global__ __launch_bounds__(32) void gemm_kernel(
    const float* __restrict__ A, int lda,
    const unsigned* __restrict__ PB,
    const float* __restrict__ bias,
    float* __restrict__ C, int ldc, int K,
    const float* __restrict__ gate, int gate_stride, int gate_off,
    int rows_per_batch) {
  const int tileM = blockIdx.x * 16;
  const int nt0 = blockIdx.y * 2;
  const int lane = threadIdx.x;
  const int half = lane >> 4;
  const int ml = lane & 15;
  const int ktiles = K >> 5;
  const float* arow = A + (size_t)(tileM + ml) * lda + 8 * half;
  v8f acc00 = {}, acc01 = {}, acc10 = {}, acc11 = {};
  for (int kk = 0; kk < K; kk += 64) {
    int k32 = kk >> 5;
    v16bf a0 = load_afrag(arow + kk);
    v16bf b00 = load_bpacked(PB, ktiles, nt0, k32, lane);
    v16bf b01 = load_bpacked(PB, ktiles, nt0 + 1, k32, lane);
    acc00 = __builtin_amdgcn_wmma_f32_16x16x32_bf16(false, a0, false, b00,
                                                    (short)0, acc00, false, false);
    acc01 = __builtin_amdgcn_wmma_f32_16x16x32_bf16(false, a0, false, b01,
                                                    (short)0, acc01, false, false);
    v16bf a1 = load_afrag(arow + kk + 32);
    v16bf b10 = load_bpacked(PB, ktiles, nt0, k32 + 1, lane);
    v16bf b11 = load_bpacked(PB, ktiles, nt0 + 1, k32 + 1, lane);
    acc10 = __builtin_amdgcn_wmma_f32_16x16x32_bf16(false, a1, false, b10,
                                                    (short)0, acc10, false, false);
    acc11 = __builtin_amdgcn_wmma_f32_16x16x32_bf16(false, a1, false, b11,
                                                    (short)0, acc11, false, false);
  }
#pragma unroll
  for (int g = 0; g < 2; ++g) {
    int col = (nt0 + g) * 16 + ml;
    float bv = bias[col];
#pragma unroll
    for (int v = 0; v < 8; ++v) {
      int row = tileM + v + 8 * half;
      float x = (g == 0 ? acc00[v] + acc10[v] : acc01[v] + acc11[v]) + bv;
      float* outp = C + (size_t)row * ldc + col;
      if (EPI == 1) x = gelu_f(x);
      if (EPI == 2) {
        int b = row / rows_per_batch;
        x = *outp + gate[(size_t)b * gate_stride + gate_off + col] * x;
      }
      *outp = x;
    }
  }
}

// ---------------------------------------------------------------------------
// Timestep embedding -> c = (silu(emb@t1+b1))@t2+b2 ; also silu(c)
// ---------------------------------------------------------------------------
__global__ __launch_bounds__(256) void timec_kernel(
    const int* __restrict__ tsteps,
    const float* __restrict__ t1w, const float* __restrict__ t1b,
    const float* __restrict__ t2w, const float* __restrict__ t2b,
    float* __restrict__ c_out, float* __restrict__ sc_out) {
  __shared__ float c1[BATCH * DMODEL];
  const int tid = threadIdx.x;
  for (int i = 0; i < 4; ++i) {
    int o = tid + 256 * i;
    int b = o >> 8, d = o & 255;
    float t = (float)tsteps[b];
    float acc = t1b[d];
    for (int k = 0; k < 256; ++k) {
      float fr, e;
      if (k < 128) { fr = __expf(-9.210340372f * (float)k / 128.0f);         e = __cosf(t * fr); }
      else         { fr = __expf(-9.210340372f * (float)(k - 128) / 128.0f); e = __sinf(t * fr); }
      acc += e * t1w[(size_t)k * 256 + d];
    }
    c1[o] = silu_f(acc);
  }
  __syncthreads();
  for (int i = 0; i < 4; ++i) {
    int o = tid + 256 * i;
    int b = o >> 8, d = o & 255;
    float acc = t2b[d];
    for (int k = 0; k < 256; ++k) acc += c1[b * 256 + k] * t2w[(size_t)k * 256 + d];
    c_out[o] = acc;
    sc_out[o] = silu_f(acc);
  }
}

// mod = silu(c) @ mod_w + mod_b   (B x 1536; tiny, plain VALU)
__global__ __launch_bounds__(256) void mod_kernel(
    const float* __restrict__ sc, const float* __restrict__ mw,
    const float* __restrict__ mb, float* __restrict__ mod) {
  int o = blockIdx.x * 256 + threadIdx.x;  // < 6144
  int b = o / 1536, col = o % 1536;
  float acc = mb[col];
  const float* s = sc + b * 256;
  for (int k = 0; k < 256; ++k) acc += s[k] * mw[(size_t)k * 1536 + col];
  mod[o] = acc;
}

// LayerNorm (+optional AdaLN): out = ln(h)*(1+sc)+s ; one wave per row
__global__ __launch_bounds__(256) void ln_kernel(
    const float* __restrict__ H, float* __restrict__ Out,
    const float* __restrict__ mod, int mod_stride, int s_off, int sc_off,
    int rows_per_batch) {
  int row = blockIdx.x * 8 + (threadIdx.x >> 5);
  int lane = threadIdx.x & 31;
  const float* hr = H + (size_t)row * DMODEL;
  float v[8];
  float sum = 0.0f;
#pragma unroll
  for (int i = 0; i < 8; ++i) { v[i] = hr[lane + 32 * i]; sum += v[i]; }
#pragma unroll
  for (int off = 16; off; off >>= 1) sum += __shfl_xor(sum, off);
  float mean = sum * (1.0f / 256.0f);
  float var = 0.0f;
#pragma unroll
  for (int i = 0; i < 8; ++i) { float d = v[i] - mean; var += d * d; }
#pragma unroll
  for (int off = 16; off; off >>= 1) var += __shfl_xor(var, off);
  float rstd = rsqrtf(var * (1.0f / 256.0f) + 1e-6f);
  int b = row / rows_per_batch;
#pragma unroll
  for (int i = 0; i < 8; ++i) {
    int c = lane + 32 * i;
    float y = (v[i] - mean) * rstd;
    if (mod)
      y = y * (1.0f + mod[(size_t)b * mod_stride + sc_off + c]) +
          mod[(size_t)b * mod_stride + s_off + c];
    Out[(size_t)row * DMODEL + c] = y;
  }
}

// attention scores + softmax: one wave per (b,h,q); each lane owns 6 keys
__global__ __launch_bounds__(32) void attn_kernel(
    const float* __restrict__ qkv, float* __restrict__ att) {
  int idx = blockIdx.x;
  int b = idx / (NHEAD * NTOK);
  int rem = idx % (NHEAD * NTOK);
  int h = rem / NTOK, q = rem % NTOK;
  int lane = threadIdx.x;
  const float* qv = qkv + ((size_t)(b * NTOK + q)) * (3 * DMODEL) + h * HDIM;
  float s[6];
#pragma unroll
  for (int t = 0; t < 6; ++t) {
    int k = lane + 32 * t;
    const float* kv = qkv + ((size_t)(b * NTOK + k)) * (3 * DMODEL) + DMODEL + h * HDIM;
    float d = 0.0f;
    for (int e = 0; e < HDIM; ++e) d += qv[e] * kv[e];
    s[t] = d * 0.125f;
  }
  float m = -3.4e38f;
#pragma unroll
  for (int t = 0; t < 6; ++t) m = fmaxf(m, s[t]);
#pragma unroll
  for (int off = 16; off; off >>= 1) m = fmaxf(m, __shfl_xor(m, off));
  float sum = 0.0f;
#pragma unroll
  for (int t = 0; t < 6; ++t) { s[t] = __expf(s[t] - m); sum += s[t]; }
#pragma unroll
  for (int off = 16; off; off >>= 1) sum += __shfl_xor(sum, off);
  float inv = fast_rcp(sum);
  float* arow = att + (((size_t)b * NHEAD + h) * NTOK + q) * NTOK;
#pragma unroll
  for (int t = 0; t < 6; ++t) arow[lane + 32 * t] = s[t] * inv;
}

// o = att @ v : one thread per output channel
__global__ __launch_bounds__(256) void av_kernel(
    const float* __restrict__ att, const float* __restrict__ qkv,
    float* __restrict__ o) {
  int row = blockIdx.x;
  int b = row / NTOK, q = row % NTOK;
  int c = threadIdx.x;
  int h = c >> 6, d = c & 63;
  const float* ar = att + (((size_t)b * NHEAD + h) * NTOK + q) * NTOK;
  float acc = 0.0f;
  for (int k = 0; k < NTOK; ++k)
    acc += ar[k] * qkv[((size_t)(b * NTOK + k)) * (3 * DMODEL) + 2 * DMODEL + h * HDIM + d];
  o[(size_t)row * DMODEL + c] = acc;
}

// ---------------------------------------------------------------------------
// Fused pairwise MLP. h j-tile (16x256 f32) staged into LDS by the Tensor
// Data Mover (2D D#: tile 256x16, stride 256, data_size=4B), then
// p0(gelu) -> p1(gelu) -> po fully on-chip with WMMA.
// ---------------------------------------------------------------------------
__global__ __launch_bounds__(256) void pair_kernel(
    const float* __restrict__ hn,
    const unsigned* __restrict__ p0p, const float* __restrict__ p0b,
    const unsigned* __restrict__ p1p, const float* __restrict__ p1b,
    const float* __restrict__ pow_, const float* __restrict__ pob,
    float* __restrict__ la) {
  __shared__ float s_hi[DMODEL];
  __shared__ float s_hj[16 * DMODEL];
  __shared__ float s_u[16 * DMODEL];
  __shared__ float s_v[16 * DMODEL];
  const int b = blockIdx.z, i = blockIdx.y, j0 = blockIdx.x * 16;
  const int tid = threadIdx.x;
  s_hi[tid] = hn[((size_t)(b * NTOK) + i) * DMODEL + tid];

  if (tid < 32) {
    // TDM: DMA rows j0..j0+15 of hn[b] (16 x 256 fp32) into s_hj.
    unsigned long long ga =
        (unsigned long long)(const void*)(hn + ((size_t)(b * NTOK) + j0) * DMODEL);
    unsigned lds = (unsigned)(size_t)(void*)&s_hj[0];
    v4u g0;
    g0[0] = 1u;                                               // count=1
    g0[1] = lds;                                              // lds_addr
    g0[2] = (unsigned)(ga & 0xFFFFFFFFu);                     // global_addr[31:0]
    g0[3] = (unsigned)((ga >> 32) & 0x1FFFFFFu) | (2u << 30); // addr[56:32]|type=2
    v8i g1;
    g1[0] = (2 << 16);          // data_size = 4 bytes
    g1[1] = (int)(256u << 16);  // tensor_dim0 = 256 (bits 79:48, low half)
    g1[2] = (int)(16u << 16);   // tensor_dim0 hi=0 | tensor_dim1 = 16
    g1[3] = (int)(256u << 16);  // tensor_dim1 hi=0 | tile_dim0 = 256
    g1[4] = 16;                 // tile_dim1 = 16, tile_dim2 = 0
    g1[5] = 256;                // tensor_dim0_stride = 256 (low 32)
    g1[6] = 0;
    g1[7] = 0;
    v4i z4 = {0, 0, 0, 0};
    v8i z8 = {0, 0, 0, 0, 0, 0, 0, 0};
    __builtin_amdgcn_tensor_load_to_lds(g0, g1, z4, z4, z8, 0);
    __builtin_amdgcn_s_wait_tensorcnt(0);
  }
  __syncthreads();

  const int wave = tid >> 5, lane = tid & 31;
  const int half = lane >> 4, ml = lane & 15;

  // stage 1: u = gelu(f @ p0w + p0b), K = 768
  for (int cti = 0; cti < 2; ++cti) {
    int ct = wave * 2 + cti;
    int col = ct * 16 + ml;
    v8f acc = {};
    for (int kk = 0; kk < 3 * DMODEL; kk += 32) {
      int seg = kk >> 8;
      int kc0 = kk & 255;
      v8u au;
#pragma unroll
      for (int v = 0; v < 8; ++v) {
        int kb = (v < 4 ? 2 * v : 16 + 2 * (v - 4)) + 8 * half;
        int kc = kc0 + kb;
        float hi0 = s_hi[kc],     hj0 = s_hj[ml * DMODEL + kc];
        float hi1 = s_hi[kc + 1], hj1 = s_hj[ml * DMODEL + kc + 1];
        float f0 = (seg == 0) ? (hi0 + hj0) : (seg == 1 ? fabsf(hi0 - hj0) : hi0 * hj0);
        float f1 = (seg == 0) ? (hi1 + hj1) : (seg == 1 ? fabsf(hi1 - hj1) : hi1 * hj1);
        au[v] = pack_bf16(f0, f1);
      }
      v16bf af = __builtin_bit_cast(v16bf, au);
      v16bf bfv = load_bpacked(p0p, (3 * DMODEL) >> 5, ct, kk >> 5, lane);
      acc = __builtin_amdgcn_wmma_f32_16x16x32_bf16(false, af, false, bfv,
                                                    (short)0, acc, false, false);
    }
    float bias = p0b[col];
#pragma unroll
    for (int v = 0; v < 8; ++v)
      s_u[(v + 8 * half) * DMODEL + col] = gelu_f(acc[v] + bias);
  }
  __syncthreads();

  // stage 2: v = gelu(u @ p1w + p1b), K = 256
  for (int cti = 0; cti < 2; ++cti) {
    int ct = wave * 2 + cti;
    int col = ct * 16 + ml;
    v8f acc = {};
    for (int kk = 0; kk < DMODEL; kk += 32) {
      v8u au;
#pragma unroll
      for (int v = 0; v < 8; ++v) {
        int kb = (v < 4 ? 2 * v : 16 + 2 * (v - 4)) + 8 * half;
        au[v] = pack_bf16(s_u[ml * DMODEL + kk + kb], s_u[ml * DMODEL + kk + kb + 1]);
      }
      v16bf af = __builtin_bit_cast(v16bf, au);
      v16bf bfv = load_bpacked(p1p, DMODEL >> 5, ct, kk >> 5, lane);
      acc = __builtin_amdgcn_wmma_f32_16x16x32_bf16(false, af, false, bfv,
                                                    (short)0, acc, false, false);
    }
    float bias = p1b[col];
#pragma unroll
    for (int v = 0; v < 8; ++v)
      s_v[(v + 8 * half) * DMODEL + col] = gelu_f(acc[v] + bias);
  }
  __syncthreads();

  // stage 3: logits = v @ po_w + po_b ; mask diag ; scale by 1/TEMP
  if (tid < 16) {
    int j = j0 + tid;
    float acc = pob[0];
    for (int k = 0; k < DMODEL; ++k) acc += s_v[tid * DMODEL + k] * pow_[k];
    float logit = (i == j) ? -1.0e9f : acc;
    la[(((size_t)b * NTOK) + i) * NTOK + j] = logit * INV_TEMP;
  }
}

// Sinkhorn row normalize; one wave per row
__global__ __launch_bounds__(32) void sink_row(float* __restrict__ la) {
  int row = blockIdx.x;
  float* p = la + (size_t)row * NTOK;
  int lane = threadIdx.x;
  float v[6];
  float m = -3.4e38f;
#pragma unroll
  for (int t = 0; t < 6; ++t) { v[t] = p[lane + 32 * t]; m = fmaxf(m, v[t]); }
#pragma unroll
  for (int off = 16; off; off >>= 1) m = fmaxf(m, __shfl_xor(m, off));
  float s = 0.0f;
#pragma unroll
  for (int t = 0; t < 6; ++t) s += __expf(v[t] - m);
#pragma unroll
  for (int off = 16; off; off >>= 1) s += __shfl_xor(s, off);
  float lse = m + __logf(s);
#pragma unroll
  for (int t = 0; t < 6; ++t) p[lane + 32 * t] = v[t] - lse;
}

// Sinkhorn column normalize; one wave per column
__global__ __launch_bounds__(32) void sink_col(float* __restrict__ la) {
  int idx = blockIdx.x;
  int b = idx / NTOK, j = idx % NTOK;
  float* base = la + (size_t)b * NTOK * NTOK + j;
  int lane = threadIdx.x;
  float v[6];
  float m = -3.4e38f;
#pragma unroll
  for (int t = 0; t < 6; ++t) { v[t] = base[(size_t)(lane + 32 * t) * NTOK]; m = fmaxf(m, v[t]); }
#pragma unroll
  for (int off = 16; off; off >>= 1) m = fmaxf(m, __shfl_xor(m, off));
  float s = 0.0f;
#pragma unroll
  for (int t = 0; t < 6; ++t) s += __expf(v[t] - m);
#pragma unroll
  for (int off = 16; off; off >>= 1) s += __shfl_xor(s, off);
  float lse = m + __logf(s);
#pragma unroll
  for (int t = 0; t < 6; ++t) base[(size_t)(lane + 32 * t) * NTOK] = v[t] - lse;
}

// x0 = 0.5*(exp(la) + exp(la)^T)
__global__ __launch_bounds__(256) void sym_kernel(
    const float* __restrict__ la, float* __restrict__ x0) {
  int idx = blockIdx.x * 256 + threadIdx.x;
  int b = idx / (NTOK * NTOK);
  int r = (idx / NTOK) % NTOK;
  int c = idx % NTOK;
  float a = __expf(la[idx]);
  float bt = __expf(la[((size_t)b * NTOK + c) * NTOK + r]);
  x0[idx] = 0.5f * (a + bt);
}

// ---------------------------------------------------------------------------
// Host orchestration. Input index map (setup_inputs() insertion order):
// 0:x 1:timesteps 2:in_w 3:in_b 4:t1_w 5:t1_b 6:t2_w 7:t2_b
// 8+10L .. : blocks[L] {mod_w,mod_b,qkv_w,qkv_b,o_w,o_b,m1_w,m1_b,m2_w,m2_b}
// 48:p0_w 49:p0_b 50:p1_w 51:p1_b 52:po_w 53:po_b
// ---------------------------------------------------------------------------
static inline int repack_blocks(int K, int N) { return ((N / 16) * (K / 32) * 32 + 255) / 256; }

extern "C" void kernel_launch(void* const* d_in, const int* in_sizes, int n_in,
                              void* d_out, int out_size, void* d_ws, size_t ws_size,
                              hipStream_t stream) {
  (void)in_sizes; (void)n_in; (void)out_size; (void)ws_size;
  const float* x      = (const float*)d_in[0];
  const int*   tstep  = (const int*)d_in[1];
  const float* in_w   = (const float*)d_in[2];
  const float* in_b   = (const float*)d_in[3];
  const float* t1_w   = (const float*)d_in[4];
  const float* t1_b   = (const float*)d_in[5];
  const float* t2_w   = (const float*)d_in[6];
  const float* t2_b   = (const float*)d_in[7];
  const float* p0_w   = (const float*)d_in[48];
  const float* p0_b   = (const float*)d_in[49];
  const float* p1_w   = (const float*)d_in[50];
  const float* p1_b   = (const float*)d_in[51];
  const float* po_w   = (const float*)d_in[52];
  const float* po_b   = (const float*)d_in[53];

  float* W = (float*)d_ws;
  float* h     = W;                 W += BN * DMODEL;
  float* cbuf  = W;                 W += BATCH * DMODEL;
  float* scbuf = W;                 W += BATCH * DMODEL;
  float* modb  = W;                 W += BATCH * 6 * DMODEL;
  float* abuf  = W;                 W += BN * DMODEL;
  float* qkvb  = W;                 W += BN * 3 * DMODEL;
  float* attb  = W;                 W += BATCH * NHEAD * NTOK * NTOK;
  float* obuf  = W;                 W += BN * DMODEL;
  float* m1buf = W;                 W += BN * 4 * DMODEL;

  // bf16-packed weights (dwords = K*N/2 each)
  unsigned* U = (unsigned*)W;
  unsigned* pin = U;                U += (INPUT_DIM * DMODEL) / 2;
  unsigned* pqkv[NLAYER]; unsigned* pow2[NLAYER];
  unsigned* pm1[NLAYER];  unsigned* pm2[NLAYER];
  for (int L = 0; L < NLAYER; ++L) {
    pqkv[L] = U;  U += (DMODEL * 3 * DMODEL) / 2;
    pow2[L] = U;  U += (DMODEL * DMODEL) / 2;
    pm1[L]  = U;  U += (DMODEL * 4 * DMODEL) / 2;
    pm2[L]  = U;  U += (4 * DMODEL * DMODEL) / 2;
  }
  unsigned* pp0 = U;                U += (3 * DMODEL * DMODEL) / 2;
  unsigned* pp1 = U;                U += (DMODEL * DMODEL) / 2;

  float* la = (float*)d_out;
  float* x0 = la + BATCH * NTOK * NTOK;

  dim3 wv(32);

  // ---- weight repacking (independent of activations) ----
  repack_kernel<<<repack_blocks(INPUT_DIM, DMODEL), 256, 0, stream>>>(in_w, pin, INPUT_DIM, DMODEL);
  for (int L = 0; L < NLAYER; ++L) {
    repack_kernel<<<repack_blocks(DMODEL, 3 * DMODEL), 256, 0, stream>>>(
        (const float*)d_in[8 + 10 * L + 2], pqkv[L], DMODEL, 3 * DMODEL);
    repack_kernel<<<repack_blocks(DMODEL, DMODEL), 256, 0, stream>>>(
        (const float*)d_in[8 + 10 * L + 4], pow2[L], DMODEL, DMODEL);
    repack_kernel<<<repack_blocks(DMODEL, 4 * DMODEL), 256, 0, stream>>>(
        (const float*)d_in[8 + 10 * L + 6], pm1[L], DMODEL, 4 * DMODEL);
    repack_kernel<<<repack_blocks(4 * DMODEL, DMODEL), 256, 0, stream>>>(
        (const float*)d_in[8 + 10 * L + 8], pm2[L], 4 * DMODEL, DMODEL);
  }
  repack_kernel<<<repack_blocks(3 * DMODEL, DMODEL), 256, 0, stream>>>(p0_w, pp0, 3 * DMODEL, DMODEL);
  repack_kernel<<<repack_blocks(DMODEL, DMODEL), 256, 0, stream>>>(p1_w, pp1, DMODEL, DMODEL);

  timec_kernel<<<1, 256, 0, stream>>>(tstep, t1_w, t1_b, t2_w, t2_b, cbuf, scbuf);

  // h = x @ in_w + in_b
  gemm_kernel<0><<<dim3(BN / 16, DMODEL / 32), wv, 0, stream>>>(
      x, INPUT_DIM, pin, in_b, h, DMODEL, INPUT_DIM, nullptr, 0, 0, 0);

  for (int L = 0; L < NLAYER; ++L) {
    const float* mod_w = (const float*)d_in[8 + 10 * L + 0];
    const float* mod_b = (const float*)d_in[8 + 10 * L + 1];
    const float* qkv_b = (const float*)d_in[8 + 10 * L + 3];
    const float* o_b   = (const float*)d_in[8 + 10 * L + 5];
    const float* m1_b  = (const float*)d_in[8 + 10 * L + 7];
    const float* m2_b  = (const float*)d_in[8 + 10 * L + 9];

    mod_kernel<<<24, 256, 0, stream>>>(scbuf, mod_w, mod_b, modb);
    ln_kernel<<<BN / 8, 256, 0, stream>>>(h, abuf, modb, 6 * DMODEL, 0, DMODEL, NTOK);
    gemm_kernel<0><<<dim3(BN / 16, (3 * DMODEL) / 32), wv, 0, stream>>>(
        abuf, DMODEL, pqkv[L], qkv_b, qkvb, 3 * DMODEL, DMODEL, nullptr, 0, 0, 0);
    attn_kernel<<<BATCH * NHEAD * NTOK, 32, 0, stream>>>(qkvb, attb);
    av_kernel<<<BN, 256, 0, stream>>>(attb, qkvb, obuf);
    gemm_kernel<2><<<dim3(BN / 16, DMODEL / 32), wv, 0, stream>>>(
        obuf, DMODEL, pow2[L], o_b, h, DMODEL, DMODEL,
        modb, 6 * DMODEL, 2 * DMODEL, NTOK);
    ln_kernel<<<BN / 8, 256, 0, stream>>>(h, abuf, modb, 6 * DMODEL, 3 * DMODEL,
                                          4 * DMODEL, NTOK);
    gemm_kernel<1><<<dim3(BN / 16, (4 * DMODEL) / 32), wv, 0, stream>>>(
        abuf, DMODEL, pm1[L], m1_b, m1buf, 4 * DMODEL, DMODEL, nullptr, 0, 0, 0);
    gemm_kernel<2><<<dim3(BN / 16, DMODEL / 32), wv, 0, stream>>>(
        m1buf, 4 * DMODEL, pm2[L], m2_b, h, DMODEL, 4 * DMODEL,
        modb, 6 * DMODEL, 5 * DMODEL, NTOK);
  }

  ln_kernel<<<BN / 8, 256, 0, stream>>>(h, abuf, nullptr, 0, 0, 0, NTOK);

  pair_kernel<<<dim3(NTOK / 16, NTOK, BATCH), 256, 0, stream>>>(
      abuf, pp0, p0_b, pp1, p1_b, po_w, po_b, la);

  for (int it = 0; it < SINK_ITERS; ++it) {
    sink_row<<<BN, 32, 0, stream>>>(la);
    sink_col<<<BN, 32, 0, stream>>>(la);
  }

  sym_kernel<<<(BATCH * NTOK * NTOK) / 256, 256, 0, stream>>>(la, x0);
}
